// Model_30245159698931
// MI455X (gfx1250) — matrix operand order repaired and checked by reading, not compiled, not run on hardware
//
#include <hip/hip_runtime.h>
#include <hip/hip_bf16.h>

typedef __attribute__((ext_vector_type(16))) _Float16 v16h;
typedef __attribute__((ext_vector_type(8)))  float    v8f;
typedef __attribute__((ext_vector_type(4)))  unsigned int u32x4;
typedef __attribute__((ext_vector_type(8)))  int      i32x8;
typedef __attribute__((ext_vector_type(4)))  int      i32x4;

#define NN 50000
#define NE 800000

#if defined(__has_builtin)
#if __has_builtin(__builtin_amdgcn_tensor_load_to_lds) && __has_builtin(__builtin_amdgcn_s_wait_tensorcnt)
#define USE_TDM 1
#endif
#endif

#ifdef USE_TDM
// DMA a 1-D tensor of `nelem` f32 from global `W` into LDS at `ldsPtr` via the Tensor Data Mover.
// D# per CDNA5 ISA 8.3/8.4: count=1, type=2, data_size=2 (4B), no multicast/pad/iterate.
static __device__ inline void tdm_load_f32(const float* W, void* ldsPtr, unsigned nelem) {
    unsigned lds_off = (unsigned)(uintptr_t)ldsPtr;  // low 32 bits of generic = LDS offset
    unsigned long long ga = (unsigned long long)(uintptr_t)W;
    u32x4 g0 = { 1u, lds_off, (unsigned)ga,
                 ((unsigned)(ga >> 32) & 0x01FFFFFFu) | 0x80000000u };
    i32x8 g1 = { (int)0x00020000,            // data_size=2 (4B)
                 (int)(nelem << 16),         // tensor_dim0 low 16 -> bits 63:48
                 (int)0x00010000,            // tensor_dim1 = 1
                 (int)(nelem << 16),         // tile_dim0
                 1,                          // tile_dim1 = 1
                 (int)nelem, 0, 0 };         // tensor_dim0_stride
    i32x4 z4 = { 0, 0, 0, 0 };
#if __clang_major__ >= 23
    i32x8 z8 = { 0, 0, 0, 0, 0, 0, 0, 0 };
    __builtin_amdgcn_tensor_load_to_lds(g0, g1, z4, z4, z8, 0);
#else
    __builtin_amdgcn_tensor_load_to_lds(g0, g1, z4, z4, 0);
#endif
    __builtin_amdgcn_s_wait_tensorcnt(0);
}
#endif

// ---------------- WMMA fragment helpers (wave32, 16x16x32 f16) ----------------
static __device__ inline v16h load_a_frag(const _Float16* __restrict__ t, int ld, int lane) {
    int m = lane & 15;
    int g = lane >> 4;
    v16h a;
#pragma unroll
    for (int i = 0; i < 8; ++i) {
        int k = ((i < 4) ? (2 * i) : (16 + 2 * (i - 4))) + (g << 3);
        a[2 * i]     = t[m * ld + k];
        a[2 * i + 1] = t[m * ld + k + 1];
    }
    return a;
}
// B fragments pre-swizzled in LDS: [frag][lane][16 halves] (32B/lane contiguous -> 2x ds_load_b128)
static __device__ inline v16h load_b_pref(const _Float16* __restrict__ sBf, int frag, int lane) {
    return *(const v16h*)(sBf + (size_t)frag * 512 + lane * 16);
}
static __device__ inline v8f wmma_f16(v16h a, v16h b, v8f c) {
    return __builtin_amdgcn_wmma_f32_16x16x32_f16(false, a, false, b, (short)0, c, false, false);
}
// Stage W (dout x ldw, row-major f32) into B-fragment layout.
static __device__ inline void stage_b(_Float16* sBf, const float* __restrict__ W,
                                      int ldw, int dout, int KS, int NB, int tid) {
    int total = KS * NB * 512;
    for (int idx = tid; idx < total; idx += 128) {
        int j    = idx & 15;
        int lane = (idx >> 4) & 31;
        int frag = idx >> 9;
        int nb   = frag % NB;
        int ks   = frag / NB;
        int n = nb * 16 + (lane & 15);
        int k = ks * 32 + ((lane >> 4) << 3) + ((j < 8) ? j : (8 + j));
        sBf[idx] = (n < dout) ? (_Float16)W[(size_t)n * ldw + k] : (_Float16)0.0f;
    }
}

// ---------------- kernel 1: zero accumulators ----------------
__global__ void k_zero(float* p, long n) {
    long i = (long)blockIdx.x * blockDim.x + threadIdx.x;
    if (i < n) p[i] = 0.0f;
}

// ---------------- kernel 2: degree = segment_sum(1, dst) ----------------
__global__ void k_deg(const int* __restrict__ dst, float* __restrict__ deg) {
    int i = blockIdx.x * blockDim.x + threadIdx.x;
    if (i < NE) atomicAdd(&deg[(unsigned)dst[i]], 1.0f);
}

// ---------------- kernel 3: m1 = (2e)@Wm111^T + b ; scatter-add to h_acc[dst] ----------------
__global__ void k_m1_scatter(const float* __restrict__ e, const int* __restrict__ dst,
                             const float* __restrict__ W, const float* __restrict__ bias,
                             float* __restrict__ h_acc) {
    __shared__ __align__(32) _Float16 sBf[2 * 512];
    __shared__ float    sb[32];
    __shared__ __align__(32) _Float16 sA[4][16 * 32];
    int tid = threadIdx.x;
    __builtin_prefetch(W, 0, 1);
    stage_b(sBf, W, 32, 32, 1, 2, tid);
    if (tid < 32) sb[tid] = bias[tid];

    int wave = tid >> 5, lane = tid & 31;
    int g = lane >> 4, row = lane & 15;
    unsigned eb = (unsigned)(blockIdx.x * 4 + wave) * 16u;
    {
        const float4* ep = (const float4*)(e + (eb + row) * 32u + (unsigned)(g << 4));
        v16h av;
#pragma unroll
        for (int q = 0; q < 4; ++q) {
            float4 v = ep[q];
            av[4 * q + 0] = (_Float16)(2.0f * v.x);
            av[4 * q + 1] = (_Float16)(2.0f * v.y);
            av[4 * q + 2] = (_Float16)(2.0f * v.z);
            av[4 * q + 3] = (_Float16)(2.0f * v.w);
        }
        *(v16h*)&sA[wave][row * 32 + (g << 4)] = av;
    }
    __syncthreads();

    v16h a = load_a_frag(&sA[wave][0], 32, lane);
    int n = lane & 15;
#pragma unroll
    for (int nb = 0; nb < 2; ++nb) {
        v16h b = load_b_pref(sBf, nb, lane);
        float bv = sb[nb * 16 + n];
        v8f c;
#pragma unroll
        for (int r = 0; r < 8; ++r) c[r] = bv;
        c = wmma_f16(a, b, c);
#pragma unroll
        for (int r = 0; r < 8; ++r) {
            int m = (g << 3) + r;
            unsigned d = (unsigned)dst[eb + m];
            atomicAdd(&h_acc[d * 32u + (unsigned)(nb * 16 + n)], c[r]);
        }
    }
}

// ---------------- kernel 4: h = relu(h_acc/denom) -> f16 ----------------
__global__ void k_h_relu(const float* __restrict__ h_acc, const float* __restrict__ deg,
                         _Float16* __restrict__ h16) {
    unsigned i = blockIdx.x * blockDim.x + threadIdx.x;
    if (i < (unsigned)NN * 32u) {
        float d = fmaxf(deg[i >> 5], 1.0f);
        h16[i] = (_Float16)fmaxf(h_acc[i] / d, 0.0f);
    }
}

// ---------------- kernel 5: m2 = [h[src], e]@Wm^T + bm ; scatter to hn_acc[dst] ----------------
__global__ void k_m2_scatter(const float* __restrict__ e, const _Float16* __restrict__ h16,
                             const int* __restrict__ src, const int* __restrict__ dst,
                             const float* __restrict__ W, const float* __restrict__ bias,
                             float* __restrict__ hn_acc) {
    __shared__ __align__(32) _Float16 sBf[16 * 512];  // 16 KB
    __shared__ float    sb[128];
    __shared__ __align__(32) _Float16 sA[4][16 * 64]; // 8 KB
#ifdef USE_TDM
    __shared__ __align__(16) float sWraw[64 * 128];   // 32 KB staged by the TDM
#endif
    int tid = threadIdx.x;
    __builtin_prefetch(W, 0, 1);
#ifdef USE_TDM
    if (tid < 32) tdm_load_f32(W, sWraw, 8192u);
    __syncthreads();
    stage_b(sBf, sWraw, 64, 128, 2, 8, tid);
#else
    stage_b(sBf, W, 64, 128, 2, 8, tid);
#endif
    if (tid < 128) sb[tid] = bias[tid];

    int wave = tid >> 5, lane = tid & 31;
    int g = lane >> 4, row = lane & 15;
    unsigned eb = (unsigned)(blockIdx.x * 4 + wave) * 16u;
    unsigned edge = eb + row;
    if (g == 0) {          // cols 0..31 = h[src]
        const v16h* hp = (const v16h*)(h16 + (unsigned)src[edge] * 32u);
        *(v16h*)&sA[wave][row * 64 + 0]  = hp[0];
        *(v16h*)&sA[wave][row * 64 + 16] = hp[1];
    } else {               // cols 32..63 = e
        const float4* ep = (const float4*)(e + edge * 32u);
#pragma unroll
        for (int h = 0; h < 2; ++h) {
            v16h av;
#pragma unroll
            for (int q = 0; q < 4; ++q) {
                float4 v = ep[4 * h + q];
                av[4 * q + 0] = (_Float16)v.x; av[4 * q + 1] = (_Float16)v.y;
                av[4 * q + 2] = (_Float16)v.z; av[4 * q + 3] = (_Float16)v.w;
            }
            *(v16h*)&sA[wave][row * 64 + 32 + 16 * h] = av;
        }
    }
    __syncthreads();

    int n = lane & 15;
#pragma unroll
    for (int nb = 0; nb < 8; ++nb) {
        float bv = sb[nb * 16 + n];
        v8f c;
#pragma unroll
        for (int r = 0; r < 8; ++r) c[r] = bv;
#pragma unroll
        for (int ks = 0; ks < 2; ++ks) {
            v16h a = load_a_frag(&sA[wave][ks * 32], 64, lane);
            v16h b = load_b_pref(sBf, ks * 8 + nb, lane);
            c = wmma_f16(a, b, c);
        }
#pragma unroll
        for (int r = 0; r < 8; ++r) {
            int m = (g << 3) + r;
            unsigned d = (unsigned)dst[eb + m];
            atomicAdd(&hn_acc[d * 128u + (unsigned)(nb * 16 + n)], c[r]);
        }
    }
}

// ---------------- kernel 6: h2 = relu((2*hn_acc/denom)@Wap^T + bap) ----------------
__global__ void k_h2(const float* __restrict__ hn_acc, const float* __restrict__ deg,
                     const float* __restrict__ W, const float* __restrict__ bias,
                     float* __restrict__ h2) {
    __shared__ __align__(32) _Float16 sBf[32 * 512];   // 32 KB
    __shared__ float    sb[128];
    __shared__ __align__(32) _Float16 sA[4][16 * 128]; // 16 KB
    int tid = threadIdx.x;
    __builtin_prefetch(W, 0, 1);
    stage_b(sBf, W, 128, 128, 4, 8, tid);
    if (tid < 128) sb[tid] = bias[tid];

    int wave = tid >> 5, lane = tid & 31;
    int g = lane >> 4, row = lane & 15;
    int tile = blockIdx.x * 4 + wave;
    bool active = tile < (NN / 16);
    unsigned nodeBase = active ? (unsigned)tile * 16u : 0u;
    unsigned node = nodeBase + row;
    float d = fmaxf(deg[node], 1.0f);
    float sc = 2.0f / d;
    {
        const float4* hp = (const float4*)(hn_acc + node * 128u + (unsigned)(g << 6));
#pragma unroll
        for (int h = 0; h < 4; ++h) {
            v16h av;
#pragma unroll
            for (int q = 0; q < 4; ++q) {
                float4 v = hp[4 * h + q];
                av[4 * q + 0] = (_Float16)(sc * v.x); av[4 * q + 1] = (_Float16)(sc * v.y);
                av[4 * q + 2] = (_Float16)(sc * v.z); av[4 * q + 3] = (_Float16)(sc * v.w);
            }
            *(v16h*)&sA[wave][row * 128 + (g << 6) + 16 * h] = av;
        }
    }
    __syncthreads();

    int n = lane & 15;
#pragma unroll
    for (int nb = 0; nb < 8; ++nb) {
        float bv = sb[nb * 16 + n];
        v8f c;
#pragma unroll
        for (int r = 0; r < 8; ++r) c[r] = bv;
#pragma unroll
        for (int ks = 0; ks < 4; ++ks) {
            v16h a = load_a_frag(&sA[wave][ks * 32], 128, lane);
            v16h b = load_b_pref(sBf, ks * 8 + nb, lane);
            c = wmma_f16(a, b, c);
        }
        if (active) {
#pragma unroll
            for (int r = 0; r < 8; ++r) {
                int m = (g << 3) + r;
                h2[(nodeBase + m) * 128u + (unsigned)(nb * 16 + n)] = fmaxf(c[r], 0.0f);
            }
        }
    }
}

// ---------------- kernel 7: eh = e@Wfc111^T + b1 ; cc = 10*(h2[src]+h2[dst]);
//                  score = [cc, eh]@Wfc^T + bfc  (K=256, N padded 8->16) ----------------
__global__ void k_final(const float* __restrict__ e, const float* __restrict__ h2,
                        const int* __restrict__ src, const int* __restrict__ dst,
                        const float* __restrict__ W1, const float* __restrict__ b1,
                        const float* __restrict__ W2, const float* __restrict__ b2,
                        float* __restrict__ out) {
    __shared__ __align__(32) _Float16 sB1f[8 * 512];   // 8 KB  (Wfc111, KS=1, NB=8)
    __shared__ float    sb1[128];
    __shared__ __align__(32) _Float16 sB2f[8 * 512];   // 8 KB  (Wfc, KS=8, NB=1, dout=8 padded)
    __shared__ __align__(32) _Float16 sE[4][16 * 32];  // 4 KB
    __shared__ __align__(32) _Float16 sX[4][16 * 256]; // 32 KB concat([cc, eh])
#ifdef USE_TDM
    __shared__ __align__(16) float sW2raw[8 * 256];    // 8 KB staged by the TDM
#endif
    int tid = threadIdx.x;
    __builtin_prefetch(W1, 0, 1);
#ifdef USE_TDM
    if (tid < 32) tdm_load_f32(W2, sW2raw, 2048u);
    __syncthreads();
    stage_b(sB2f, sW2raw, 256, 8, 8, 1, tid);
#else
    stage_b(sB2f, W2, 256, 8, 8, 1, tid);
#endif
    stage_b(sB1f, W1, 32, 128, 1, 8, tid);
    if (tid < 128) sb1[tid] = b1[tid];

    int wave = tid >> 5, lane = tid & 31;
    int g = lane >> 4, row = lane & 15;
    unsigned eb = (unsigned)(blockIdx.x * 4 + wave) * 16u;
    unsigned edge = eb + row;
    {
        const float4* ep = (const float4*)(e + edge * 32u + (unsigned)(g << 4));
        v16h av;
#pragma unroll
        for (int q = 0; q < 4; ++q) {
            float4 v = ep[q];
            av[4 * q + 0] = (_Float16)v.x; av[4 * q + 1] = (_Float16)v.y;
            av[4 * q + 2] = (_Float16)v.z; av[4 * q + 3] = (_Float16)v.w;
        }
        *(v16h*)&sE[wave][row * 32 + (g << 4)] = av;
    }
    // cc part of concat: cols 0..127
    {
        const float4* sp = (const float4*)(h2 + (unsigned)src[edge] * 128u + (unsigned)(g << 6));
        const float4* dp = (const float4*)(h2 + (unsigned)dst[edge] * 128u + (unsigned)(g << 6));
#pragma unroll
        for (int h = 0; h < 4; ++h) {
            float4 a0 = sp[4 * h + 0], c0 = dp[4 * h + 0];
            float4 a1 = sp[4 * h + 1], c1 = dp[4 * h + 1];
            float4 a2 = sp[4 * h + 2], c2 = dp[4 * h + 2];
            float4 a3 = sp[4 * h + 3], c3 = dp[4 * h + 3];
            v16h av;
            av[0]  = (_Float16)(10.0f * (a0.x + c0.x)); av[1]  = (_Float16)(10.0f * (a0.y + c0.y));
            av[2]  = (_Float16)(10.0f * (a0.z + c0.z)); av[3]  = (_Float16)(10.0f * (a0.w + c0.w));
            av[4]  = (_Float16)(10.0f * (a1.x + c1.x)); av[5]  = (_Float16)(10.0f * (a1.y + c1.y));
            av[6]  = (_Float16)(10.0f * (a1.z + c1.z)); av[7]  = (_Float16)(10.0f * (a1.w + c1.w));
            av[8]  = (_Float16)(10.0f * (a2.x + c2.x)); av[9]  = (_Float16)(10.0f * (a2.y + c2.y));
            av[10] = (_Float16)(10.0f * (a2.z + c2.z)); av[11] = (_Float16)(10.0f * (a2.w + c2.w));
            av[12] = (_Float16)(10.0f * (a3.x + c3.x)); av[13] = (_Float16)(10.0f * (a3.y + c3.y));
            av[14] = (_Float16)(10.0f * (a3.z + c3.z)); av[15] = (_Float16)(10.0f * (a3.w + c3.w));
            *(v16h*)&sX[wave][row * 256 + (g << 6) + 16 * h] = av;
        }
    }
    __syncthreads();

    int n = lane & 15;
    // eh = e @ Wfc111^T + b1 -> concat cols 128..255
    {
        v16h a = load_a_frag(&sE[wave][0], 32, lane);
#pragma unroll
        for (int nb = 0; nb < 8; ++nb) {
            v16h b = load_b_pref(sB1f, nb, lane);
            float bv = sb1[nb * 16 + n];
            v8f c;
#pragma unroll
            for (int r = 0; r < 8; ++r) c[r] = bv;
            c = wmma_f16(a, b, c);
#pragma unroll
            for (int r = 0; r < 8; ++r) {
                int m = (g << 3) + r;
                sX[wave][m * 256 + 128 + nb * 16 + n] = (_Float16)c[r];
            }
        }
    }
    __syncthreads();

    // score = X @ Wfc^T + bfc   (K=256 -> 8-WMMA chain)
    {
        float bv = (n < 8) ? b2[n] : 0.0f;
        v8f c;
#pragma unroll
        for (int r = 0; r < 8; ++r) c[r] = bv;
#pragma unroll
        for (int ks = 0; ks < 8; ++ks) {
            v16h a = load_a_frag(&sX[wave][ks * 32], 256, lane);
            v16h b = load_b_pref(sB2f, ks, lane);
            c = wmma_f16(a, b, c);
        }
        if (n < 8) {
#pragma unroll
            for (int r = 0; r < 8; ++r) {
                int m = (g << 3) + r;
                out[(eb + m) * 8u + (unsigned)n] = c[r];
            }
        }
    }
}

static inline size_t al256(size_t x) { return (x + 255) & ~(size_t)255; }

extern "C" void kernel_launch(void* const* d_in, const int* in_sizes, int n_in,
                              void* d_out, int out_size, void* d_ws, size_t ws_size,
                              hipStream_t stream) {
    const int*   src    = (const int*)d_in[0];
    const int*   dst    = (const int*)d_in[1];
    // d_in[2] = nfeats (only its shape is used by the reference)
    const float* efeats = (const float*)d_in[3];
    const float* Wm111  = (const float*)d_in[4];
    const float* bm111  = (const float*)d_in[5];
    const float* Wm     = (const float*)d_in[6];
    const float* bm     = (const float*)d_in[7];
    const float* Wap    = (const float*)d_in[8];
    const float* bap    = (const float*)d_in[9];
    const float* Wfc111 = (const float*)d_in[10];
    const float* bfc111 = (const float*)d_in[11];
    const float* Wfc    = (const float*)d_in[12];
    const float* bfc    = (const float*)d_in[13];
    float* out = (float*)d_out;

    char* ws = (char*)d_ws;
    size_t o_deg  = 0;
    size_t o_hacc = al256(o_deg  + (size_t)NN * sizeof(float));
    size_t o_hn   = al256(o_hacc + (size_t)NN * 32 * sizeof(float));
    size_t o_h16  = al256(o_hn   + (size_t)NN * 128 * sizeof(float));
    size_t o_h2   = al256(o_h16  + (size_t)NN * 32 * sizeof(_Float16));

    float*    deg    = (float*)(ws + o_deg);
    float*    h_acc  = (float*)(ws + o_hacc);
    float*    hn_acc = (float*)(ws + o_hn);
    _Float16* h16    = (_Float16*)(ws + o_h16);
    float*    h2     = (float*)(ws + o_h2);

    long nz = (long)(o_h16 / sizeof(float));
    k_zero<<<(int)((nz + 255) / 256), 256, 0, stream>>>((float*)ws, nz);
    k_deg<<<(NE + 255) / 256, 256, 0, stream>>>(dst, deg);
    k_m1_scatter<<<NE / 64, 128, 0, stream>>>(efeats, dst, Wm111, bm111, h_acc);
    k_h_relu<<<(NN * 32 + 255) / 256, 256, 0, stream>>>(h_acc, deg, h16);
    k_m2_scatter<<<NE / 64, 128, 0, stream>>>(efeats, h16, src, dst, Wm, bm, hn_acc);
    k_h2<<<(NN / 16 + 3) / 4, 128, 0, stream>>>(hn_acc, deg, Wap, bap, h2);
    k_final<<<NE / 64, 128, 0, stream>>>(efeats, h2, src, dst, Wfc111, bfc111, Wfc, bfc, out);
    (void)in_sizes; (void)n_in; (void)out_size; (void)ws_size;
}
